// RCNNNet_56453050139224
// MI455X (gfx1250) — matrix operand order repaired and checked by reading, not compiled, not run on hardware
//
#include <hip/hip_runtime.h>
#include <hip/hip_bf16.h>
#include <math.h>
#include <stdint.h>

// ---------------------------------------------------------------------------
// RCNNNet for MI455X (gfx1250, wave32, WMMA).
// All dense math runs through v_wmma_f32_16x16x32_f16 (f16 in, f32 accum).
// SA stages fuse ball-query + gather + 3-layer MLP + maxpool in LDS.
// Attention fuses QK^T + column softmax + att^T*V per 16-query tile in LDS.
// Hot GEMM paths are chunk-uniform (no per-element selects), use float4
// global loads and 16B packed LDS stores, and double-buffer the A tile.
// Workspace requirement: 3 * 65536*128 floats = ~96 MB.
// ---------------------------------------------------------------------------

typedef _Float16 v16h __attribute__((ext_vector_type(16)));
typedef _Float16 h8   __attribute__((ext_vector_type(8)));
typedef float    v8f  __attribute__((ext_vector_type(8)));

#define BATCH 128
#define NPTS0 512

// A-fragment K offset for 16-bit A (16x32, MxK), contiguous order k = 16*kh + e (matches B).
__device__ __forceinline__ int koffA(int e, int kh) {
  return kh * 16 + e;
}

// 8 consecutive floats; float4 pair when 16B-aligned (uniform per call site).
__device__ __forceinline__ void load8(const float* p, float* v) {
  if ((((uintptr_t)p) & 15) == 0) {
    const float4* q = (const float4*)p;
    float4 a = q[0], b = q[1];
    v[0] = a.x; v[1] = a.y; v[2] = a.z; v[3] = a.w;
    v[4] = b.x; v[5] = b.y; v[6] = b.z; v[7] = b.w;
  } else {
#pragma unroll
    for (int i = 0; i < 8; ++i) v[i] = p[i];
  }
}

// ---------------------------------------------------------------------------
// Generic GEMM: Y[M,O] = act( [X1 | X2][M,K1+K2] * W[O,Ktot]^T + bias )
// Block = 256 threads = 8 waves; block tile 64x32; one 16x16 WMMA tile/wave.
// A tile double-buffered in LDS. Fast chunks (fully in X1 or X2) do pure
// vector loads; only boundary chunks pay per-element select cost.
// Row/col out-of-range lanes use clamped addresses; store mask handles them.
// ---------------------------------------------------------------------------
__global__ __launch_bounds__(256) void gemm_kernel(
    const float* __restrict__ X1, int K1, int sx1,
    const float* __restrict__ X2, int K2, int sx2,
    const float* __restrict__ W, const float* __restrict__ bias,
    float* __restrict__ Y, int M, int O, int act)
{
  __shared__ __align__(16) _Float16 As[2][64 * 40];  // 64 x 32 k, stride 40
  int tid = threadIdx.x, wave = tid >> 5, lane = tid & 31, kh = lane >> 4;
  int mbase = blockIdx.x * 64;
  int wm = (wave >> 1) * 16;
  int n0 = blockIdx.y * 32 + (wave & 1) * 16;
  int Ktot = K1 + K2;
  int ocol = n0 + (lane & 15);
  int oc = ocol < O ? ocol : O - 1;             // clamp; store mask covers rest
  const float* wrow = W + (size_t)oc * Ktot;
  // cooperative A-fill: each thread fills 8 consecutive k of one row
  int frow = tid >> 2;                          // 0..63
  int fk = (tid & 3) * 8;                       // 0,8,16,24
  int mrowf = mbase + frow;
  int mcf = mrowf < M ? mrowf : M - 1;          // clamp; store mask covers rest
  int lr = wm + (lane & 15);

  auto fillA = [&](int kb, _Float16* buf) {
    float v[8];
    if (kb + 32 <= K1) {                        // chunk fully in X1 (fast)
      load8(X1 + (size_t)mcf * sx1 + kb + fk, v);
    } else if (kb >= K1 && kb + 32 <= Ktot) {   // chunk fully in X2 (fast)
      load8(X2 + (size_t)mcf * sx2 + (kb - K1) + fk, v);
    } else {                                    // boundary chunk (rare)
#pragma unroll
      for (int i = 0; i < 8; ++i) {
        int k = kb + fk + i;
        int i1 = k < K1 ? k : K1 - 1;
        int t2 = k - K1;
        t2 = t2 < 0 ? 0 : t2;
        if (K2 > 0 && t2 >= K2) t2 = K2 - 1;
        bool use2 = (k >= K1) && (K2 > 0);
        const float* p = use2 ? X2 + (size_t)mcf * sx2 + t2
                              : X1 + (size_t)mcf * sx1 + i1;
        float val = *p;
        v[i] = (k < Ktot) ? val : 0.f;          // K-tail must be exact zeros
      }
    }
    h8 hv;
#pragma unroll
    for (int i = 0; i < 8; ++i) hv[i] = (_Float16)v[i];
    *(h8*)(buf + frow * 40 + fk) = hv;          // one 16B LDS store
  };

  int nch = (Ktot + 31) >> 5;
  v8f acc = {};
  fillA(0, As[0]);
  __syncthreads();
  for (int ci = 0; ci < nch; ++ci) {
    int kb = ci << 5;
    const _Float16* buf = As[ci & 1];
    v16h a, b;
#pragma unroll
    for (int e = 0; e < 16; ++e)
      a[e] = buf[lr * 40 + koffA(e, kh)];
    float wv[16];
    if (kb + 32 <= Ktot) {                      // full chunk: vector W loads
      load8(wrow + kb + kh * 16, wv);
      load8(wrow + kb + kh * 16 + 8, wv + 8);
    } else {                                    // tail: clamp only (A is 0)
#pragma unroll
      for (int e = 0; e < 16; ++e) {
        int k = kb + kh * 16 + e;
        wv[e] = wrow[k < Ktot ? k : Ktot - 1];
      }
    }
#pragma unroll
    for (int e = 0; e < 16; ++e) b[e] = (_Float16)wv[e];
    if (kb + 32 < Ktot)
      __builtin_prefetch(wrow + kb + 32);       // global_prefetch_b8
    if (ci + 1 < nch) fillA((ci + 1) << 5, As[(ci + 1) & 1]);
    acc = __builtin_amdgcn_wmma_f32_16x16x32_f16(false, a, false, b,
                                                 (short)0, acc, false, false);
    __syncthreads();
  }
  float bv = bias[oc];
#pragma unroll
  for (int r = 0; r < 8; ++r) {
    int row = mbase + wm + r + 8 * kh;
    if (row < M && ocol < O) {
      float v = acc[r] + bv;
      if (act) v = fmaxf(v, 0.f);
      Y[(size_t)row * O + ocol] = v;
    }
  }
}

// ---------------------------------------------------------------------------
// Fused self-attention for one (batch, 16-query tile).
// ---------------------------------------------------------------------------
__global__ __launch_bounds__(256) void attention_kernel(
    const float* __restrict__ f, float* __restrict__ out,
    int N, int C, float scale)
{
  __shared__ __align__(16) _Float16 fm_sh[16 * 256];
  __shared__ float s_sh[512 * 16];
  __shared__ float red[256];
  __shared__ float colmax[16], colsum[16];
  int tid = threadIdx.x, wave = tid >> 5, lane = tid & 31, kh = lane >> 4;
  int bb = blockIdx.y;
  int m0 = blockIdx.x * 16;
  const float* fb = f + (size_t)bb * N * C;

  for (int t = tid; t < 16 * C; t += 256)
    fm_sh[t] = (_Float16)fb[(size_t)(m0 + t / C) * C + (t % C)];
  __syncthreads();

  // ---- scores: S[n][m] = f[n].f[m0+m] * scale ----
  for (int n0 = wave * 16; n0 < N; n0 += 128) {
    v8f acc = {};
    int m = lane & 15;
    int ncol = n0 + (lane & 15);
    for (int kb = 0; kb < C; kb += 32) {
      v16h a, b;
#pragma unroll
      for (int e = 0; e < 16; ++e)
        a[e] = fm_sh[m * C + kb + koffA(e, kh)];
      float kv[16];
      load8(fb + (size_t)ncol * C + kb + kh * 16, kv);      // vector key loads
      load8(fb + (size_t)ncol * C + kb + kh * 16 + 8, kv + 8);
#pragma unroll
      for (int e = 0; e < 16; ++e) b[e] = (_Float16)kv[e];
      acc = __builtin_amdgcn_wmma_f32_16x16x32_f16(false, a, false, b,
                                                   (short)0, acc, false, false);
    }
#pragma unroll
    for (int r = 0; r < 8; ++r)
      s_sh[ncol * 16 + (r + 8 * kh)] = acc[r] * scale;      // s_sh[n][m]
  }
  __syncthreads();

  // ---- column softmax over n (16 groups x 16 threads) ----
  int g = tid >> 4, l = tid & 15;
  float pm = -3.4e38f;
  for (int n = l; n < N; n += 16) pm = fmaxf(pm, s_sh[n * 16 + g]);
  red[tid] = pm;
  __syncthreads();
  if (l == 0) {
    float m = red[g * 16];
    for (int i = 1; i < 16; ++i) m = fmaxf(m, red[g * 16 + i]);
    colmax[g] = m;
  }
  __syncthreads();
  float cm = colmax[g], ps = 0.f;
  for (int n = l; n < N; n += 16) ps += __expf(s_sh[n * 16 + g] - cm);
  red[tid] = ps;
  __syncthreads();
  if (l == 0) {
    float s = 0.f;
    for (int i = 0; i < 16; ++i) s += red[g * 16 + i];
    colsum[g] = s;
  }
  __syncthreads();
  float inv = 1.f / colsum[g];
  for (int n = l; n < N; n += 16)
    s_sh[n * 16 + g] = __expf(s_sh[n * 16 + g] - cm) * inv;
  __syncthreads();

  // ---- out = att^T (16xN) @ f (NxC) ----
  float* ob = out + (size_t)bb * N * C;
  for (int ct = wave; ct < (C >> 4); ct += 8) {
    int c0 = ct * 16;
    v8f acc = {};
    int m = lane & 15;
    int col = c0 + (lane & 15);
    for (int nb = 0; nb < N; nb += 32) {
      v16h a, b;
#pragma unroll
      for (int e = 0; e < 16; ++e)
        a[e] = (_Float16)s_sh[(nb + koffA(e, kh)) * 16 + m];
#pragma unroll
      for (int e = 0; e < 16; ++e)
        b[e] = (_Float16)fb[(size_t)(nb + kh * 16 + e) * C + col];
      acc = __builtin_amdgcn_wmma_f32_16x16x32_f16(false, a, false, b,
                                                   (short)0, acc, false, false);
    }
#pragma unroll
    for (int r = 0; r < 8; ++r)
      ob[(size_t)(m0 + r + 8 * kh) * C + col] = acc[r];
  }
}

// ---------------------------------------------------------------------------
// Farthest point sampling, one block per batch (lax.scan semantics).
// ---------------------------------------------------------------------------
__global__ __launch_bounds__(256) void fps_kernel(
    const float* __restrict__ xyz, int Npts, int npoint,
    float* __restrict__ new_xyz)
{
  __shared__ float dist[512];
  __shared__ float rv[256];
  __shared__ int ri[256];
  __shared__ int sfar;
  int b = blockIdx.x, tid = threadIdx.x;
  const float* xb = xyz + (size_t)b * Npts * 3;
  for (int p = tid; p < Npts; p += 256) dist[p] = 1e10f;
  if (tid == 0) sfar = 0;
  __syncthreads();
  for (int i = 0; i < npoint; ++i) {
    int far = sfar;
    if (tid < 3)
      new_xyz[((size_t)b * npoint + i) * 3 + tid] = xb[far * 3 + tid];
    float cx = xb[far * 3], cy = xb[far * 3 + 1], cz = xb[far * 3 + 2];
    float bv = -1.f; int bi = 0;
    for (int p = tid; p < Npts; p += 256) {
      float dx = xb[p * 3] - cx, dy = xb[p * 3 + 1] - cy, dz = xb[p * 3 + 2] - cz;
      float d = fminf(dist[p], dx * dx + dy * dy + dz * dz);
      dist[p] = d;
      if (d > bv) { bv = d; bi = p; }
    }
    rv[tid] = bv; ri[tid] = bi;
    __syncthreads();
    for (int s = 128; s > 0; s >>= 1) {
      if (tid < s) {
        if (rv[tid + s] > rv[tid] ||
            (rv[tid + s] == rv[tid] && ri[tid + s] < ri[tid])) {
          rv[tid] = rv[tid + s]; ri[tid] = ri[tid + s];
        }
      }
      __syncthreads();
    }
    if (tid == 0) sfar = ri[0];
    __syncthreads();
  }
}

// ---------------------------------------------------------------------------
// In-LDS MLP layer: H(64xO) = relu( A(64xK) @ W(OxKw)^T + b ), A/H f16 in LDS.
// A's K-pad region is zero, so tail W loads need only an address clamp.
// ---------------------------------------------------------------------------
__device__ __forceinline__ void lds_mlp_layer(
    const _Float16* A, int lda, int Kpad, int Kw,
    const float* __restrict__ W, const float* __restrict__ bias,
    int O, _Float16* Hout, int ldo, int tid)
{
  int wave = tid >> 5, lane = tid & 31, kh = lane >> 4;
  int total = 4 * (O >> 4);
  for (int t = wave; t < total; t += 8) {
    int m0 = (t & 3) * 16;
    int o0 = (t >> 2) * 16;
    int m = m0 + (lane & 15);
    int o = o0 + (lane & 15);
    const float* wrow = W + (size_t)o * Kw;
    v8f acc = {};
    for (int kb = 0; kb < Kpad; kb += 32) {
      v16h a, b;
#pragma unroll
      for (int e = 0; e < 16; ++e)
        a[e] = A[m * lda + kb + koffA(e, kh)];
      float wv[16];
      if (kb + 32 <= Kw) {
        load8(wrow + kb + kh * 16, wv);
        load8(wrow + kb + kh * 16 + 8, wv + 8);
      } else {
#pragma unroll
        for (int e = 0; e < 16; ++e) {
          int k = kb + kh * 16 + e;
          wv[e] = wrow[k < Kw ? k : Kw - 1];    // clamp; A pad is zero
        }
      }
#pragma unroll
      for (int e = 0; e < 16; ++e) b[e] = (_Float16)wv[e];
      acc = __builtin_amdgcn_wmma_f32_16x16x32_f16(false, a, false, b,
                                                   (short)0, acc, false, false);
    }
    float bv = bias[o];
#pragma unroll
    for (int r = 0; r < 8; ++r)
      Hout[(m0 + r + 8 * kh) * ldo + o] = (_Float16)fmaxf(acc[r] + bv, 0.f);
  }
}

// ---------------------------------------------------------------------------
// Fused set-abstraction: one block per (batch, centroid).
// ---------------------------------------------------------------------------
__global__ __launch_bounds__(256) void sa_kernel(
    const float* __restrict__ xyz, const float* __restrict__ feats,
    int Npts, int Cf,
    const float* __restrict__ new_xyz, int npoint, float r2,
    const float* __restrict__ W1, const float* __restrict__ b1, int C1, int Kw1,
    const float* __restrict__ W2, const float* __restrict__ b2, int C2,
    const float* __restrict__ W3, const float* __restrict__ b3, int C3,
    float* __restrict__ out)
{
  __shared__ __align__(16) _Float16 G[64 * 160];  // gathered / layer-2 out
  __shared__ __align__(16) _Float16 H[64 * 256];  // layer-1 / layer-3 out
  __shared__ unsigned mw[16];
  __shared__ int sel[64];
  __shared__ float cen[3];
  int tid = threadIdx.x;
  int j = blockIdx.x % npoint, b = blockIdx.x / npoint;
  const float* xb = xyz + (size_t)b * Npts * 3;
  const float* fbp = feats + (size_t)b * Npts * Cf;
  if (tid < 3) cen[tid] = new_xyz[((size_t)b * npoint + j) * 3 + tid];
  if (tid < 16) mw[tid] = 0u;
  __syncthreads();
  float cx = cen[0], cy = cen[1], cz = cen[2];

  for (int p = tid; p < Npts; p += 256) {
    float dx = xb[p * 3] - cx, dy = xb[p * 3 + 1] - cy, dz = xb[p * 3 + 2] - cz;
    if (dx * dx + dy * dy + dz * dz < r2)
      atomicOr(&mw[p >> 5], 1u << (p & 31));
  }
  __syncthreads();
  if (tid == 0) {                       // ordered selection of first 64 hits
    int cnt = 0, nw = Npts >> 5;
    for (int w = 0; w < nw && cnt < 64; ++w) {
      unsigned word = mw[w];
      while (word && cnt < 64) {
        int bit = __builtin_ctz(word);
        sel[cnt++] = (w << 5) + bit;
        word &= word - 1;
      }
    }
    int first = (cnt > 0) ? sel[0] : 0;
    while (cnt < 64) sel[cnt++] = first;
  }
  __syncthreads();

  const int KP = 160;                   // 131 padded to multiple of 32
  for (int t = tid; t < 64 * 3; t += 256) {
    int r = t / 3, c = t % 3;
    G[r * KP + c] = (_Float16)(xb[sel[r] * 3 + c] - cen[c]);
  }
  for (int t = tid; t < 64 * Cf; t += 256) {
    int r = t / Cf, c = t % Cf;
    G[r * KP + 3 + c] = (_Float16)fbp[(size_t)sel[r] * Cf + c];
  }
  int padc = KP - 3 - Cf;
  for (int t = tid; t < 64 * padc; t += 256) {
    int r = t / padc, c = t % padc;
    G[r * KP + 3 + Cf + c] = (_Float16)0.f;
  }
  __syncthreads();

  lds_mlp_layer(G, KP, KP, Kw1, W1, b1, C1, H, C1, tid);
  __syncthreads();
  lds_mlp_layer(H, C1, C1, C1, W2, b2, C2, G, C2, tid);
  __syncthreads();
  lds_mlp_layer(G, C2, C2, C2, W3, b3, C3, H, C3, tid);
  __syncthreads();

  for (int c = tid; c < C3; c += 256) {
    float m = -3.4e38f;
    for (int r = 0; r < 64; ++r) m = fmaxf(m, (float)H[r * C3 + c]);
    out[((size_t)b * npoint + j) * C3 + c] = m;
  }
}

// ---------------------------------------------------------------------------
__global__ void maxpool_kernel(const float* __restrict__ X, float* __restrict__ Y,
                               int groups, int C, int total)
{
  int i = blockIdx.x * 256 + threadIdx.x;
  if (i >= total) return;
  int r = i / C, c = i % C;
  float m = -3.4e38f;
  for (int g = 0; g < groups; ++g)
    m = fmaxf(m, X[((size_t)r * groups + g) * C + c]);
  Y[i] = m;
}

// ---------------------------------------------------------------------------
extern "C" void kernel_launch(void* const* d_in, const int* in_sizes, int n_in,
                              void* d_out, int out_size, void* d_ws, size_t ws_size,
                              hipStream_t stream)
{
  (void)in_sizes; (void)n_in; (void)out_size; (void)ws_size;
  const float* xyz     = (const float*)d_in[0];
  const float* reflect = (const float*)d_in[1];
  const float* tmask   = (const float*)d_in[2];
  const float* xuW1 = (const float*)d_in[3];  const float* xub1 = (const float*)d_in[4];
  const float* xuW2 = (const float*)d_in[5];  const float* xub2 = (const float*)d_in[6];
  const float* fuW1 = (const float*)d_in[7];  const float* fub1 = (const float*)d_in[8];
  const float* fuW2 = (const float*)d_in[9];  const float* fub2 = (const float*)d_in[10];
  const float* mgW  = (const float*)d_in[11]; const float* mgb  = (const float*)d_in[12];
  const float* s1W1 = (const float*)d_in[13]; const float* s1b1 = (const float*)d_in[14];
  const float* s1W2 = (const float*)d_in[15]; const float* s1b2 = (const float*)d_in[16];
  const float* s1W3 = (const float*)d_in[17]; const float* s1b3 = (const float*)d_in[18];
  const float* s2W1 = (const float*)d_in[19]; const float* s2b1 = (const float*)d_in[20];
  const float* s2W2 = (const float*)d_in[21]; const float* s2b2 = (const float*)d_in[22];
  const float* s2W3 = (const float*)d_in[23]; const float* s2b3 = (const float*)d_in[24];
  const float* s3W1 = (const float*)d_in[25]; const float* s3b1 = (const float*)d_in[26];
  const float* s3W2 = (const float*)d_in[27]; const float* s3b2 = (const float*)d_in[28];
  const float* s3W3 = (const float*)d_in[29]; const float* s3b3 = (const float*)d_in[30];
  const float* cW1  = (const float*)d_in[31]; const float* cb1  = (const float*)d_in[32];
  const float* cW2  = (const float*)d_in[33]; const float* cb2  = (const float*)d_in[34];
  const float* cW3  = (const float*)d_in[35]; const float* cb3  = (const float*)d_in[36];
  const float* rW1  = (const float*)d_in[37]; const float* rb1  = (const float*)d_in[38];
  const float* rW2  = (const float*)d_in[39]; const float* rb2  = (const float*)d_in[40];
  const float* rW3  = (const float*)d_in[41]; const float* rb3  = (const float*)d_in[42];
  float* outp = (float*)d_out;

  const int M0 = BATCH * NPTS0;                 // 65536
  const size_t BIG = (size_t)M0 * 128;          // 8,388,608 floats
  float* w  = (float*)d_ws;
  float* A0 = w;
  float* A1 = w + BIG;
  float* A2 = w + 2 * BIG;
  // overlays on A0 (free after attention-1) and A1 (free after SA1):
  float* nx1 = A0;                              // 128*128*3
  float* f1  = nx1 + 49152;                     // 128*128*128
  float* f1a = f1 + 2097152;
  float* nx2 = f1a + 2097152;                   // 128*32*3
  float* f2  = nx2 + 12288;                     // 128*32*256
  float* f2a = f2 + 1048576;
  float* h3a = f2a + 1048576;                   // 4096*256
  float* h3b = A1;                              // 4096*256
  float* h3c = h3b + 1048576;                   // 4096*512
  float* f3  = h3c + 2097152;                   // 128*512
  float* hh1 = f3 + 65536;                      // 128*256
  float* hh2 = hh1 + 32768;                     // 128*256

  auto gemm = [&](const float* X1, int K1, int sx1,
                  const float* X2, int K2, int sx2,
                  const float* W, const float* bias,
                  float* Y, int M, int O, int act) {
    dim3 grid((M + 63) / 64, (O + 31) / 32);
    gemm_kernel<<<grid, 256, 0, stream>>>(X1, K1, sx1, X2, K2, sx2,
                                          W, bias, Y, M, O, act);
  };

  // xyz_up: 3 -> 128 -> 128
  gemm(xyz, 3, 3, nullptr, 0, 0, xuW1, xub1, A0, M0, 128, 1);
  gemm(A0, 128, 128, nullptr, 0, 0, xuW2, xub2, A1, M0, 128, 1);     // ux
  // feat_up: [reflect|mask] 2 -> 128 -> 128
  gemm(reflect, 1, 1, tmask, 1, 1, fuW1, fub1, A0, M0, 128, 1);
  gemm(A0, 128, 128, nullptr, 0, 0, fuW2, fub2, A2, M0, 128, 1);     // uf
  // merge: [ux|uf] 256 -> 128
  gemm(A1, 128, 128, A2, 128, 128, mgW, mgb, A0, M0, 128, 1);        // f0

  // attention 1 (N=512, C=128): f0(A0) -> A1
  attention_kernel<<<dim3(NPTS0 / 16, BATCH), 256, 0, stream>>>(
      A0, A1, NPTS0, 128, 1.f / sqrtf(128.f));

  // SA1: fps(512->128), fused group-MLP 131->128->128->128, pool 64
  fps_kernel<<<BATCH, 256, 0, stream>>>(xyz, NPTS0, 128, nx1);
  sa_kernel<<<BATCH * 128, 256, 0, stream>>>(
      xyz, A1, NPTS0, 128, nx1, 128, 0.2f * 0.2f,
      s1W1, s1b1, 128, 131, s1W2, s1b2, 128, s1W3, s1b3, 128, f1);

  // attention 2 (N=128, C=128)
  attention_kernel<<<dim3(128 / 16, BATCH), 256, 0, stream>>>(
      f1, f1a, 128, 128, 1.f / sqrtf(128.f));

  // SA2: fps(128->32), fused group-MLP 131->128->128->256
  fps_kernel<<<BATCH, 256, 0, stream>>>(nx1, 128, 32, nx2);
  sa_kernel<<<BATCH * 32, 256, 0, stream>>>(
      nx1, f1a, 128, 128, nx2, 32, 0.4f * 0.4f,
      s2W1, s2b1, 128, 131, s2W2, s2b2, 128, s2W3, s2b3, 256, f2);

  // attention 3 (N=32, C=256)
  attention_kernel<<<dim3(32 / 16, BATCH), 256, 0, stream>>>(
      f2, f2a, 32, 256, 1.f / sqrtf(256.f));

  // SA3 (global): [xyz|feats] 259 -> 256 -> 256 -> 512, maxpool over 32
  gemm(nx2, 3, 3, f2a, 256, 256, s3W1, s3b1, h3a, BATCH * 32, 256, 1);
  gemm(h3a, 256, 256, nullptr, 0, 0, s3W2, s3b2, h3b, BATCH * 32, 256, 1);
  gemm(h3b, 256, 256, nullptr, 0, 0, s3W3, s3b3, h3c, BATCH * 32, 512, 1);
  maxpool_kernel<<<(BATCH * 512 + 255) / 256, 256, 0, stream>>>(
      h3c, f3, 32, 512, BATCH * 512);

  // cls head: 512 -> 256 -> 256 -> 1  (out[0..127])
  gemm(f3, 512, 512, nullptr, 0, 0, cW1, cb1, hh1, BATCH, 256, 1);
  gemm(hh1, 256, 256, nullptr, 0, 0, cW2, cb2, hh2, BATCH, 256, 1);
  gemm(hh2, 256, 256, nullptr, 0, 0, cW3, cb3, outp, BATCH, 1, 0);
  // reg head: 512 -> 256 -> 256 -> 46 (out[128..])
  gemm(f3, 512, 512, nullptr, 0, 0, rW1, rb1, hh1, BATCH, 256, 1);
  gemm(hh1, 256, 256, nullptr, 0, 0, rW2, rb2, hh2, BATCH, 256, 1);
  gemm(hh2, 256, 256, nullptr, 0, 0, rW3, rb3, outp + BATCH, BATCH, 46, 0);
}
